// AttentionFlowLayer_2611340116451
// MI455X (gfx1250) — matrix operand order repaired and checked by reading, compiled, not run on hardware
//
#include <hip/hip_runtime.h>
#include <cstdint>

// ---------------------------------------------------------------------------
// Types
// ---------------------------------------------------------------------------
typedef __attribute__((ext_vector_type(16))) __bf16 v16bf;
typedef __attribute__((ext_vector_type(8)))  __bf16 v8bf;
typedef __attribute__((ext_vector_type(4)))  __bf16 v4bf;
typedef __attribute__((ext_vector_type(8)))  float  v8f;

union V8F { v8f v; float f[8]; };

__device__ __forceinline__ v16bf cat8(v8bf lo, v8bf hi) {
  return __builtin_shufflevector(lo, hi, 0,1,2,3,4,5,6,7,8,9,10,11,12,13,14,15);
}

__device__ __forceinline__ v8f wmma_bf16(v16bf a, v16bf b, v8f c) {
  // D = A(16x32 bf16) * B(32x16 bf16) + C(16x16 f32)
  return __builtin_amdgcn_wmma_f32_16x16x32_bf16(
      /*neg_a=*/false, a, /*neg_b=*/false, b,
      /*c_mod=*/(short)0, c, /*reuse_a=*/false, /*reuse_b=*/false);
}

// Async memory -> LDS copy (16 bytes per lane), tracked by ASYNCcnt.
// lds_off is a byte offset relative to the workgroup's LDS allocation.
__device__ __forceinline__ void async_load16(uint32_t lds_off, const void* gptr) {
  asm volatile("global_load_async_to_lds_b128 %0, %1, off"
               :: "v"(lds_off), "v"(gptr) : "memory");
}

__device__ __forceinline__ void wait_async0() {
  asm volatile("s_wait_asynccnt 0" ::: "memory");
}

// Problem constants
#define BB   8
#define CS   1024
#define QS   128
#define DD   1024
#define H8   4096
#define MM   (BB * CS)   // 8192 rows of cat2 / out

// ---------------------------------------------------------------------------
// K1: beta_w f32 -> bf16   (4096*4096 elements, 4 per thread)
// ---------------------------------------------------------------------------
__global__ void k_cvt_beta(const float* __restrict__ w, __bf16* __restrict__ o) {
  size_t t = (size_t)blockIdx.x * blockDim.x + threadIdx.x;
  const float4 v = ((const float4*)w)[t];
  v4bf r = { (__bf16)v.x, (__bf16)v.y, (__bf16)v.z, (__bf16)v.w };
  *(v4bf*)(o + 4 * t) = r;
}

// ---------------------------------------------------------------------------
// K2a: per (b,j): qdot = dot(query_row, w_q); emit query bf16 (row-major)
//      and queryT bf16 (d-major, for the query_hat GEMM B operand)
// ---------------------------------------------------------------------------
__global__ __launch_bounds__(256)
void k_qprep(const float* __restrict__ query, const float* __restrict__ alpha_w,
             float* __restrict__ qdot, __bf16* __restrict__ qbf,
             __bf16* __restrict__ qTbf) {
  int row = blockIdx.x;            // b*128 + j
  int b = row >> 7, j = row & 127;
  int t = threadIdx.x;
  const float* q  = query + (size_t)row * DD;
  const float* wq = alpha_w + DD;
  float s = 0.f;
#pragma unroll
  for (int e = 0; e < 4; e++) {
    int d = t + 256 * e;
    float v = q[d];
    s += v * wq[d];
    qbf[(size_t)row * DD + d] = (__bf16)v;
    qTbf[((size_t)(b * DD + d)) * QS + j] = (__bf16)v;
  }
#pragma unroll
  for (int m = 16; m >= 1; m >>= 1) s += __shfl_xor(s, m, 32);
  __shared__ float red[8];
  if ((t & 31) == 0) red[t >> 5] = s;
  __syncthreads();
  if (t == 0) {
    float tot = 0.f;
    for (int q2 = 0; q2 < 8; q2++) tot += red[q2];
    qdot[row] = tot;
  }
}

// ---------------------------------------------------------------------------
// K2b: per (b,i): cdot = dot(context_row, w_c); emit (context*w_cq) as bf16
// ---------------------------------------------------------------------------
__global__ __launch_bounds__(256)
void k_cprep(const float* __restrict__ context, const float* __restrict__ alpha_w,
             float* __restrict__ cdot, __bf16* __restrict__ ctxwbf) {
  int row = blockIdx.x;            // b*1024 + i
  int t = threadIdx.x;
  const float* c   = context + (size_t)row * DD;
  const float* wc  = alpha_w;
  const float* wcq = alpha_w + 2 * DD;
  float s = 0.f;
#pragma unroll
  for (int e = 0; e < 4; e++) {
    int d = t + 256 * e;
    float v = c[d];
    s += v * wc[d];
    ctxwbf[(size_t)row * DD + d] = (__bf16)(v * wcq[d]);
  }
#pragma unroll
  for (int m = 16; m >= 1; m >>= 1) s += __shfl_xor(s, m, 32);
  __shared__ float red[8];
  if ((t & 31) == 0) red[t >> 5] = s;
  __syncthreads();
  if (t == 0) {
    float tot = 0.f;
    for (int q2 = 0; q2 < 8; q2++) tot += red[q2];
    cdot[row] = tot;
  }
}

// ---------------------------------------------------------------------------
// K3: S tile (16 i-rows x 128 j) via WMMA bf16, K=1024, fused row-softmax.
//     Emits c_q_w (bf16, softmax weights) and rmax (f32, max_j S).
//     Block = 128 threads (4 waves); wave w owns j in [32w, 32w+32).
// ---------------------------------------------------------------------------
__global__ __launch_bounds__(128)
void k_S(const __bf16* __restrict__ ctxwbf, const __bf16* __restrict__ qbf,
         const float* __restrict__ cdot, const float* __restrict__ qdot,
         const float* __restrict__ alpha_b, __bf16* __restrict__ cqw,
         float* __restrict__ rmax) {
  __shared__ alignas(16) __bf16 Abuf[16 * DD];   // 32 KB: A strip, 16 rows x K=1024
  __shared__ float Sbuf[16 * QS];                // 8 KB
  __shared__ float s_max[16], s_sum[16];

  int tile = blockIdx.x;           // b*64 + it
  int b = tile >> 6, it = tile & 63;
  int t = threadIdx.x, lane = t & 31, w = t >> 5;
  int laneLo = lane & 15, laneHi = lane >> 4;

  // cooperative fill of the (shared) A strip
  {
    const v8bf* src = (const v8bf*)(ctxwbf + ((size_t)(b * CS + it * 16)) * DD);
    v8bf* dst = (v8bf*)Abuf;
    for (int c = t; c < 2048; c += 128) dst[c] = src[c];
  }
  __syncthreads();

  v8f acc0 = {0, 0, 0, 0, 0, 0, 0, 0};
  v8f acc1 = acc0;

  for (int k = 0; k < DD; k += 32) {
    const __bf16* ap = Abuf + laneLo * DD + k + laneHi * 8;
    v8bf alo = *(const v8bf*)ap;
    v8bf ahi = *(const v8bf*)(ap + 16);
    v16bf af = cat8(alo, ahi);

    int j0 = w * 32 + laneLo;
    const __bf16* bp0 = qbf + ((size_t)(b * QS + j0)) * DD + k + laneHi * 8;
    v16bf bf0 = cat8(*(const v8bf*)bp0, *(const v8bf*)(bp0 + 16));
    const __bf16* bp1 = qbf + ((size_t)(b * QS + j0 + 16)) * DD + k + laneHi * 8;
    v16bf bf1 = cat8(*(const v8bf*)bp1, *(const v8bf*)(bp1 + 16));

    acc0 = wmma_bf16(af, bf0, acc0);
    acc1 = wmma_bf16(af, bf1, acc1);
  }

  float ab = alpha_b[0];
#pragma unroll
  for (int nt = 0; nt < 2; nt++) {
    V8F u; u.v = nt ? acc1 : acc0;
    int col = w * 32 + nt * 16 + laneLo;
    float qd = qdot[b * QS + col];
#pragma unroll
    for (int r = 0; r < 8; r++) {
      int m = r + laneHi * 8;
      Sbuf[m * QS + col] = u.f[r] + cdot[b * CS + it * 16 + m] + qd + ab;
    }
  }
  __syncthreads();

  if (t < 16) {
    const float* Sr = Sbuf + t * QS;
    float mx = -3.4e38f;
    for (int j = 0; j < QS; j++) mx = fmaxf(mx, Sr[j]);
    float sum = 0.f;
    for (int j = 0; j < QS; j++) sum += __expf(Sr[j] - mx);
    s_max[t] = mx;
    s_sum[t] = sum;
    rmax[b * CS + it * 16 + t] = mx;   // max_j S, pre-softmax (reference semantics)
  }
  __syncthreads();

  for (int c = t; c < 16 * QS; c += 128) {
    int m = c >> 7, j = c & 127;
    float v = __expf(Sbuf[c] - s_max[m]) / s_sum[m];
    cqw[((size_t)(b * CS + it * 16 + m)) * QS + j] = (__bf16)v;
  }
}

// ---------------------------------------------------------------------------
// K4: q_c_w[b,:] = softmax_i(rmax[b,:])   (one block per batch)
// ---------------------------------------------------------------------------
__global__ __launch_bounds__(256)
void k_qcw(const float* __restrict__ rmax, float* __restrict__ qcw) {
  int b = blockIdx.x, t = threadIdx.x;
  __shared__ float red[8];
  __shared__ float s_mx, s_sum;

  float mx = -3.4e38f;
  for (int i = t; i < CS; i += 256) mx = fmaxf(mx, rmax[b * CS + i]);
#pragma unroll
  for (int m = 16; m >= 1; m >>= 1) mx = fmaxf(mx, __shfl_xor(mx, m, 32));
  if ((t & 31) == 0) red[t >> 5] = mx;
  __syncthreads();
  if (t == 0) {
    float a = red[0];
    for (int q = 1; q < 8; q++) a = fmaxf(a, red[q]);
    s_mx = a;
  }
  __syncthreads();

  float sm = 0.f;
  for (int i = t; i < CS; i += 256) sm += __expf(rmax[b * CS + i] - s_mx);
#pragma unroll
  for (int m = 16; m >= 1; m >>= 1) sm += __shfl_xor(sm, m, 32);
  if ((t & 31) == 0) red[t >> 5] = sm;
  __syncthreads();
  if (t == 0) {
    float a = 0.f;
    for (int q = 0; q < 8; q++) a += red[q];
    s_sum = a;
  }
  __syncthreads();

  for (int i = t; i < CS; i += 256)
    qcw[b * CS + i] = __expf(rmax[b * CS + i] - s_mx) / s_sum;
}

// ---------------------------------------------------------------------------
// K5: context_hat[b,d] = sum_i q_c_w[b,i] * context[b,i,d]
// ---------------------------------------------------------------------------
__global__ __launch_bounds__(256)
void k_chat(const float* __restrict__ context, const float* __restrict__ qcw,
            float* __restrict__ chat) {
  int b = blockIdx.x;
  int d = blockIdx.y * 256 + threadIdx.x;
  const float* cp = context + (size_t)b * CS * DD + d;
  float acc = 0.f;
  for (int i = 0; i < CS; i++) acc += qcw[b * CS + i] * cp[(size_t)i * DD];
  chat[b * DD + d] = acc;
}

// ---------------------------------------------------------------------------
// K6: query_hat = c_q_w @ query (WMMA bf16, K=128), fused cat2 assembly:
//     cat2 = [chat | qhat | ctx*qhat | ctx*chat] in bf16.
//     Block = 128 threads (4 waves); grid (512 tiles, 8 d-blocks of 128).
// ---------------------------------------------------------------------------
__global__ __launch_bounds__(128)
void k_qhat_cat2(const __bf16* __restrict__ cqw, const __bf16* __restrict__ qTbf,
                 const float* __restrict__ context, const float* __restrict__ chat,
                 __bf16* __restrict__ cat2) {
  int tile = blockIdx.x;           // b*64 + it
  int b = tile >> 6, it = tile & 63;
  int dby = blockIdx.y;
  int t = threadIdx.x, lane = t & 31, w = t >> 5;
  int laneLo = lane & 15, laneHi = lane >> 4;

  v8f acc0 = {0, 0, 0, 0, 0, 0, 0, 0};
  v8f acc1 = acc0;

  const __bf16* arow = cqw + ((size_t)(b * CS + it * 16 + laneLo)) * QS + laneHi * 8;
#pragma unroll
  for (int k = 0; k < QS; k += 32) {
    v16bf af = cat8(*(const v8bf*)(arow + k), *(const v8bf*)(arow + k + 16));

    int d0 = dby * 128 + w * 32 + laneLo;
    const __bf16* bp0 = qTbf + ((size_t)(b * DD + d0)) * QS + k + laneHi * 8;
    v16bf bf0 = cat8(*(const v8bf*)bp0, *(const v8bf*)(bp0 + 16));
    const __bf16* bp1 = qTbf + ((size_t)(b * DD + d0 + 16)) * QS + k + laneHi * 8;
    v16bf bf1 = cat8(*(const v8bf*)bp1, *(const v8bf*)(bp1 + 16));

    acc0 = wmma_bf16(af, bf0, acc0);
    acc1 = wmma_bf16(af, bf1, acc1);
  }

#pragma unroll
  for (int nt = 0; nt < 2; nt++) {
    V8F u; u.v = nt ? acc1 : acc0;
    int d = dby * 128 + w * 32 + nt * 16 + laneLo;
    float ch = chat[b * DD + d];
#pragma unroll
    for (int r = 0; r < 8; r++) {
      int i = it * 16 + r + laneHi * 8;
      size_t row = (size_t)(b * CS + i);
      float c  = context[row * DD + d];
      float qh = u.f[r];
      size_t base = row * H8;
      cat2[base + d]            = (__bf16)ch;
      cat2[base + DD + d]       = (__bf16)qh;
      cat2[base + 2 * DD + d]   = (__bf16)(c * qh);
      cat2[base + 3 * DD + d]   = (__bf16)(c * ch);
    }
  }
}

// ---------------------------------------------------------------------------
// K7: out = cat2(8192x4096 bf16) @ beta_w^T(4096x4096 bf16) + beta_b, f32 out.
//     Block = 256 threads (8 waves) -> 128x128 output tile.
//     K staged 32-wide into LDS with async global->LDS copies, double-buffered:
//       issue tile k+32 (ASYNCcnt), WMMA on tile k, s_wait_asynccnt 0, barrier.
//     LDS arena (single __shared__ array => static LDS offset 0):
//       buf b: A at byte 16384*b, B at byte 16384*b + 8192.  Total 32 KB.
// ---------------------------------------------------------------------------
__global__ __launch_bounds__(256)
void k_gemm(const __bf16* __restrict__ A, const __bf16* __restrict__ Bw,
            const float* __restrict__ beta_b, float* __restrict__ out) {
  __shared__ alignas(16) __bf16 smem[2 * 8192];   // 32 KB, starts at LDS offset 0

  int t = threadIdx.x, lane = t & 31, w = t >> 5;
  int wm = w >> 2, wn = w & 3;
  int laneLo = lane & 15, laneHi = lane >> 4;
  int mbase = blockIdx.x * 128;
  int nbase = blockIdx.y * 128;

  v8f acc[8];
  {
    v8f z = {0, 0, 0, 0, 0, 0, 0, 0};
#pragma unroll
    for (int q = 0; q < 8; q++) acc[q] = z;
  }

  // Issue async copies of one 128x32 A tile + 128x32 B tile into buffer bb.
  auto issue_tile = [&](int bb, int k) {
#pragma unroll
    for (int c0 = 0; c0 < 512; c0 += 256) {
      int c = c0 + t;                       // 512 chunks of 16B per operand
      int row = c >> 2, col = c & 3;
      uint32_t la = (uint32_t)bb * 16384u + (uint32_t)c * 16u;
      async_load16(la,
                   A  + ((size_t)(mbase + row)) * H8 + k + col * 8);
      async_load16(la + 8192u,
                   Bw + ((size_t)(nbase + row)) * H8 + k + col * 8);
    }
  };

  issue_tile(0, 0);
  wait_async0();
  __syncthreads();

  for (int k = 0; k < H8; k += 32) {
    int bb = (k >> 5) & 1;
    if (k + 32 < H8) issue_tile(bb ^ 1, k + 32);

    const __bf16* As = smem + bb * 8192;      // [128][32]
    const __bf16* Bs = As + 4096;             // [128][32]

    v16bf af[4];
#pragma unroll
    for (int mt = 0; mt < 4; mt++) {
      const __bf16* p = As + (wm * 64 + mt * 16 + laneLo) * 32 + laneHi * 8;
      af[mt] = cat8(*(const v8bf*)p, *(const v8bf*)(p + 16));
    }
    v16bf bfr[2];
#pragma unroll
    for (int nt = 0; nt < 2; nt++) {
      const __bf16* p = Bs + (wn * 32 + nt * 16 + laneLo) * 32 + laneHi * 8;
      bfr[nt] = cat8(*(const v8bf*)p, *(const v8bf*)(p + 16));
    }
#pragma unroll
    for (int mt = 0; mt < 4; mt++)
#pragma unroll
      for (int nt = 0; nt < 2; nt++)
        acc[mt * 2 + nt] = wmma_bf16(af[mt], bfr[nt], acc[mt * 2 + nt]);

    if (k + 32 < H8) wait_async0();
    __syncthreads();
  }

#pragma unroll
  for (int nt = 0; nt < 2; nt++) {
    int n = nbase + wn * 32 + nt * 16 + laneLo;
    float bb2 = beta_b[n];
#pragma unroll
    for (int mt = 0; mt < 4; mt++) {
      V8F u; u.v = acc[mt * 2 + nt];
#pragma unroll
      for (int r = 0; r < 8; r++) {
        int m = mbase + wm * 64 + mt * 16 + r + laneHi * 8;
        out[(size_t)m * H8 + n] = u.f[r] + bb2;
      }
    }
  }
}

// ---------------------------------------------------------------------------
// Launch
// ---------------------------------------------------------------------------
extern "C" void kernel_launch(void* const* d_in, const int* in_sizes, int n_in,
                              void* d_out, int out_size, void* d_ws, size_t ws_size,
                              hipStream_t stream) {
  const float* context = (const float*)d_in[0];
  const float* query   = (const float*)d_in[1];
  const float* alpha_w = (const float*)d_in[2];
  const float* alpha_b = (const float*)d_in[3];
  const float* beta_w  = (const float*)d_in[4];
  const float* beta_b  = (const float*)d_in[5];
  float* out = (float*)d_out;

  char* p = (char*)d_ws;
  auto alloc = [&](size_t bytes) -> char* {
    char* r = p;
    p += (bytes + 255) & ~(size_t)255;
    return r;
  };
  float* qdot   = (float*)alloc((size_t)BB * QS * 4);       // 4 KB
  float* cdot   = (float*)alloc((size_t)MM * 4);            // 32 KB
  float* rmax   = (float*)alloc((size_t)MM * 4);            // 32 KB
  float* qcw    = (float*)alloc((size_t)MM * 4);            // 32 KB
  float* chat   = (float*)alloc((size_t)BB * DD * 4);       // 32 KB
  __bf16* cqw    = (__bf16*)alloc((size_t)MM * QS * 2);     // 2 MB
  __bf16* qbf    = (__bf16*)alloc((size_t)BB * QS * DD * 2);// 2 MB
  __bf16* qTbf   = (__bf16*)alloc((size_t)BB * DD * QS * 2);// 2 MB
  __bf16* ctxwbf = (__bf16*)alloc((size_t)MM * DD * 2);     // 16 MB
  __bf16* cat2   = (__bf16*)alloc((size_t)MM * H8 * 2);     // 64 MB
  __bf16* betabf = (__bf16*)alloc((size_t)H8 * H8 * 2);     // 32 MB

  k_cvt_beta<<<(H8 * H8) / (256 * 4), 256, 0, stream>>>(beta_w, betabf);
  k_qprep<<<BB * QS, 256, 0, stream>>>(query, alpha_w, qdot, qbf, qTbf);
  k_cprep<<<MM, 256, 0, stream>>>(context, alpha_w, cdot, ctxwbf);
  k_S<<<MM / 16, 128, 0, stream>>>(ctxwbf, qbf, cdot, qdot, alpha_b, cqw, rmax);
  k_qcw<<<BB, 256, 0, stream>>>(rmax, qcw);
  k_chat<<<dim3(BB, DD / 256), 256, 0, stream>>>(context, qcw, chat);
  k_qhat_cat2<<<dim3(MM / 16, DD / 128), 128, 0, stream>>>(cqw, qTbf, context, chat, cat2);
  k_gemm<<<dim3(MM / 128, H8 / 128), 256, 0, stream>>>(cat2, betabf, beta_b, out);
}
